// GRU_4123168604726
// MI455X (gfx1250) — matrix operand order repaired and checked by reading, not compiled, not run on hardware
//
#include <hip/hip_runtime.h>

#define DEVINL __device__ __forceinline__

typedef __bf16 bf16_t;
typedef bf16_t v16bf __attribute__((ext_vector_type(16)));
typedef float  v8f   __attribute__((ext_vector_type(8)));
typedef unsigned int u32x4 __attribute__((ext_vector_type(4)));
typedef int i32x8 __attribute__((ext_vector_type(8)));
typedef int i32x4 __attribute__((ext_vector_type(4)));

union BFrag { uint4 q[2]; v16bf v; };

// Problem dims
constexpr int    Tt  = 2048;
constexpr int    N3  = 768;          // 3*U
// Workspace layout (bytes)
constexpr size_t UFRAG_OFF = 0;                    // 3*16*8*1024 = 393216
constexpr size_t WFRAG_OFF = 393216;               // 48*8*1024  = 393216
constexpr size_t XWS_OFF   = 786432;               // 2048*4*48*512 = 201326592
// x_ws layout: [t][g(4)][nt(48)][lane(32)][i(8)] bf16 (C-fragment order)

DEVINL unsigned short f2bf(float f) {
  unsigned u = __builtin_bit_cast(unsigned, f);
  u += 0x7FFFu + ((u >> 16) & 1u);                 // round-to-nearest-even
  return (unsigned short)(u >> 16);
}
DEVINL float bf2f(unsigned short h) {
  unsigned u = ((unsigned)h) << 16;
  return __builtin_bit_cast(float, u);
}
DEVINL float satf(float x) { return fminf(fmaxf(x, 0.f), 1.f); }
DEVINL float tanh_fast(float x) {
  float t = __expf(-2.0f * fabsf(x));              // in (0,1], no overflow
  float r = (1.0f - t) / (1.0f + t);
  return copysignf(r, x);
}
DEVINL v8f wmma_bf16(v16bf a, v16bf b, v8f c) {
  return __builtin_amdgcn_wmma_f32_16x16x32_bf16(false, a, false, b, (short)0, c,
                                                 false, false);
}

// ---------------------------------------------------------------------------
// Swizzle a 256x768 row-major fp32 matrix into bf16 B-fragment order:
// [nt(48)][kc(8)] frags of 1024B, per-lane 32B, per ISA 16-bit B layout:
//   element (k,n): lane = (n%16) + 16*(k%16 >= 8), vg = (k%8)/2 + 4*(k%32 >= 16), pos = k&1
// ---------------------------------------------------------------------------
__global__ void swz_kernel(const float* __restrict__ src, char* __restrict__ dst) {
  int tid = blockIdx.x * 256 + threadIdx.x;
  if (tid >= 256 * N3) return;
  int k = tid / N3, n = tid - k * N3;
  int nt = n >> 4, nn = n & 15;
  int kc = k >> 5, kk = k & 31;
  int lane = nn + (((kk & 15) >= 8) ? 16 : 0);
  int vg   = ((kk & 7) >> 1) + ((kk >= 16) ? 4 : 0);
  int pos  = kk & 1;
  size_t off = ((size_t)(nt * 8 + kc) << 10) + (size_t)lane * 32 + vg * 4 + pos * 2;
  *(unsigned short*)(dst + off) = f2bf(src[tid]);
}

// ---------------------------------------------------------------------------
// Phase 1: x_proj = x @ W + bias, written as bf16 in scan C-fragment layout.
// Grid: 1024 blocks (g = bid>>8 batch-group, t0 = (bid&255)*8), 256 threads.
// Wave tt (0..7) computes all 48 N-tiles for timestep t0+tt.
// ---------------------------------------------------------------------------
__global__ void xproj_kernel(const float* __restrict__ x,
                             const float* __restrict__ bias,
                             const char* __restrict__ wfrag,
                             char* __restrict__ xws) {
  extern __shared__ __align__(16) char lds[];      // 8 t * 8 kc * 1024 = 65536 B
  const int tid = threadIdx.x;
  const int g  = blockIdx.x >> 8;
  const int t0 = (blockIdx.x & 255) * 8;

  // ---- stage A (convert fp32 -> bf16, A-fragment order) ----
#pragma unroll 4
  for (int it = 0; it < 32; ++it) {
    int pos = it * 1024 + tid * 4;                 // 0 .. 32767
    int k  = pos & 255;
    int tt = (pos >> 8) & 7;
    int bb = pos >> 11;                            // 0..15 batch-in-group
    const float4 v = *(const float4*)(x + (((size_t)(16 * g + bb) * Tt + (t0 + tt)) << 8) + k);
    unsigned lo = (unsigned)f2bf(v.x) | ((unsigned)f2bf(v.y) << 16);
    unsigned hi = (unsigned)f2bf(v.z) | ((unsigned)f2bf(v.w) << 16);
    int kc    = k >> 5;
    int halfA = ((k & 15) >= 8) ? 1 : 0;
    int vg    = ((k & 7) >> 1) + (((k & 31) >= 16) ? 4 : 0);
    int lanep = bb + (halfA << 4);
    uint2 st; st.x = lo; st.y = hi;
    *(uint2*)(lds + (size_t)tt * 8192 + (size_t)kc * 1024 + lanep * 32 + vg * 4) = st;
  }
  __syncthreads();

  const int wave = tid >> 5;                       // == local timestep tt
  const int l    = tid & 31;
  const size_t t = (size_t)(t0 + wave);
  const char* ldsA = lds + (size_t)wave * 8192;

  for (int nt = 0; nt < 48; ++nt) {
    float bv = bias[nt * 16 + (l & 15)];
    v8f acc = {0.f, 0.f, 0.f, 0.f, 0.f, 0.f, 0.f, 0.f};
#pragma unroll
    for (int kc = 0; kc < 8; ++kc) {
      BFrag a, b;
      const char* ap = ldsA + (size_t)kc * 1024 + l * 32;
      a.q[0] = *(const uint4*)(ap);
      a.q[1] = *(const uint4*)(ap + 16);
      const char* bp = wfrag + ((size_t)(nt * 8 + kc) << 10) + l * 32;
      b.q[0] = *(const uint4*)(bp);
      b.q[1] = *(const uint4*)(bp + 16);
      acc = wmma_bf16(a.v, b.v, acc);
    }
    unsigned short o[8];
#pragma unroll
    for (int i = 0; i < 8; ++i) o[i] = f2bf(acc[i] + bv);
    uint4 st;
    st.x = (unsigned)o[0] | ((unsigned)o[1] << 16);
    st.y = (unsigned)o[2] | ((unsigned)o[3] << 16);
    st.z = (unsigned)o[4] | ((unsigned)o[5] << 16);
    st.w = (unsigned)o[6] | ((unsigned)o[7] << 16);
    *(uint4*)(xws + (((t * 4 + g) * 48 + nt) << 9) + l * 16) = st;
  }
}

// ---------------------------------------------------------------------------
// Phase 2: persistent scan. Grid = 4 WGs x 512 threads (16 waves).
// WG g owns batches 16g..16g+15; wave w owns hidden columns [16w,16w+16).
// LDS: [0,128K) Uz frags, [128K,256K) Ur frags (filled by TDM),
//      [256K,256K+8K) h buf0, [+8K,+16K) h buf1 (double buffer, 1 barrier/step)
// Uh fragments (nt=w) live in 64 VGPRs per wave for the whole scan.
// ---------------------------------------------------------------------------
__global__ void scan_kernel(const char* __restrict__ ufrag,
                            const char* __restrict__ xws,
                            float* __restrict__ out) {
  extern __shared__ __align__(16) char lds[];      // 278528 B
  const int g   = blockIdx.x;
  const int tid = threadIdx.x;
  const int w   = tid >> 5;
  const int l   = tid & 31;

#if __has_builtin(__builtin_amdgcn_tensor_load_to_lds) && __has_builtin(__builtin_amdgcn_s_wait_tensorcnt)
  // ---- TDM: one DMA of Uz+Ur (262144 B) global -> LDS, wave 0 issues ----
  if (w == 0) {
    const unsigned long long ga = (unsigned long long)(uintptr_t)ufrag;
    const unsigned ldsOff = (unsigned)(uintptr_t)(void*)lds; // addr[31:0] = LDS offset
    const unsigned N0 = 262144u >> 3;                   // 32768 x 8-byte elements
    u32x4 g0;
    g0[0] = 1u;                                         // count=1, user descriptor
    g0[1] = ldsOff;                                     // lds_addr
    g0[2] = (unsigned)(ga & 0xFFFFFFFFu);               // global_addr[31:0]
    g0[3] = ((unsigned)(ga >> 32) & 0x01FFFFFFu) | 0x80000000u; // addr[56:32] | type=2
    i32x8 g1;
    g1[0] = (int)(3u << 16);                            // data_size=8B, mask=0
    g1[1] = (int)((N0 & 0xFFFFu) << 16);                // tensor_dim0[15:0]
    g1[2] = (int)((N0 >> 16) & 0xFFFFu) | (1 << 16);    // tensor_dim0[31:16] | tensor_dim1=1
    g1[3] = (int)((N0 & 0xFFFFu) << 16);                // tile_dim0 = 32768
    g1[4] = 0;                                          // tile_dim1/2 unused
    g1[5] = (int)N0;                                    // tensor_dim0_stride[31:0]
    g1[6] = 0;                                          // stride hi, dim1_stride lo
    g1[7] = 0;
    i32x4 gz4; gz4[0] = 0; gz4[1] = 0; gz4[2] = 0; gz4[3] = 0;
    i32x8 gz8;
#pragma unroll
    for (int i = 0; i < 8; ++i) gz8[i] = 0;
    __builtin_amdgcn_tensor_load_to_lds(g0, g1, gz4, gz4, gz8, 0);
    __builtin_amdgcn_s_wait_tensorcnt(0);
  }
#else
#pragma unroll 4
  for (int it = 0; it < 32; ++it) {
    int idx = it * 512 + tid;                      // 16384 uint4 total
    ((uint4*)lds)[idx] = ((const uint4*)ufrag)[idx];
  }
#endif
  // zero both h fragment buffers (16384 B = 1024 uint4)
  {
    uint4 z; z.x = z.y = z.z = z.w = 0u;
    ((uint4*)(lds + 262144))[tid] = z;
    ((uint4*)(lds + 262144))[tid + 512] = z;
  }

  // persistent Uh fragments (gate 2, nt = w)
  BFrag uhf[8];
#pragma unroll
  for (int kc = 0; kc < 8; ++kc) {
    const char* p = ufrag + ((size_t)((2 * 16 + w) * 8 + kc) << 10) + l * 32;
    uhf[kc].q[0] = *(const uint4*)(p);
    uhf[kc].q[1] = *(const uint4*)(p + 16);
  }
  float hprev[8];
#pragma unroll
  for (int i = 0; i < 8; ++i) hprev[i] = 0.f;
  __syncthreads();

  const char* ldsUz = lds;
  const char* ldsUr = lds + 131072;
  char*       hbase = lds + 262144;                // two 8192 B h buffers

  // precompute h A-fragment scatter offsets for this wave's columns
  const int kcH  = w >> 1;
  const int kk   = ((w & 1) << 4) + (l & 15);      // k index this lane writes
  const int vgH  = ((kk & 7) >> 1) + ((kk >= 16) ? 4 : 0);
  const int posH = kk & 1;
  const int laneB = ((l >> 4) << 3) + ((((kk & 15) >= 8) ? 1 : 0) << 4);
  const int hwOff = kcH * 1024 + vgH * 4 + posH * 2;

  for (int t = 0; t < Tt; ++t) {
    const char* bufR = hbase + ((t & 1) << 13);        // h(t-1)
    char*       bufW = hbase + (((t + 1) & 1) << 13);  // h(t)

    const size_t xbase = (((size_t)t * 4 + g) * 48 + w) * 512 + l * 16;
    uint4 xz4 = *(const uint4*)(xws + xbase);
    uint4 xr4 = *(const uint4*)(xws + xbase + 16 * 512);
    uint4 xh4 = *(const uint4*)(xws + xbase + 32 * 512);
    if (t + 8 < Tt) {                              // gfx1250 global_prefetch_b8
      const size_t pb = (((size_t)(t + 8) * 4 + g) * 48 + w) * 512 + l * 16;
      __builtin_prefetch(xws + pb, 0, 3);
      __builtin_prefetch(xws + pb + 16 * 512, 0, 3);
      __builtin_prefetch(xws + pb + 32 * 512, 0, 3);
    }

    v8f aZ = {0.f,0.f,0.f,0.f,0.f,0.f,0.f,0.f};
    v8f aR = aZ, aH = aZ;
#pragma unroll
    for (int kc = 0; kc < 8; ++kc) {
      BFrag a, bz, br;
      const char* ap = bufR + kc * 1024 + l * 32;
      a.q[0] = *(const uint4*)(ap);
      a.q[1] = *(const uint4*)(ap + 16);
      const char* zp = ldsUz + ((size_t)(w * 8 + kc) << 10) + l * 32;
      bz.q[0] = *(const uint4*)(zp);
      bz.q[1] = *(const uint4*)(zp + 16);
      const char* rp = ldsUr + ((size_t)(w * 8 + kc) << 10) + l * 32;
      br.q[0] = *(const uint4*)(rp);
      br.q[1] = *(const uint4*)(rp + 16);
      aZ = wmma_bf16(a.v, bz.v, aZ);
      aR = wmma_bf16(a.v, br.v, aR);
      aH = wmma_bf16(a.v, uhf[kc].v, aH);
    }

    const unsigned short* xz = (const unsigned short*)&xz4;
    const unsigned short* xr = (const unsigned short*)&xr4;
    const unsigned short* xh = (const unsigned short*)&xh4;
    float hn[8];
#pragma unroll
    for (int i = 0; i < 8; ++i) {
      float z  = satf(0.2f * (bf2f(xz[i]) + aZ[i]) + 0.5f);
      float r  = satf(0.2f * (bf2f(xr[i]) + aR[i]) + 0.5f);
      float hh = tanh_fast(bf2f(xh[i]) + r * aH[i]);
      hn[i] = z * hprev[i] + (1.f - z) * hh;
      hprev[i] = hn[i];
    }

    char* hwr = bufW + hwOff;
#pragma unroll
    for (int i = 0; i < 8; ++i)
      *(unsigned short*)(hwr + (laneB + i) * 32) = f2bf(hn[i]);
    __syncthreads();                               // h(t) visible; old reads done
  }

  // h_last -> d_out (B=64, U=256) fp32
#pragma unroll
  for (int i = 0; i < 8; ++i) {
    int row = 16 * g + ((l >> 4) << 3) + i;
    out[(size_t)row * 256 + (w << 4) + (l & 15)] = hprev[i];
  }
}

// ---------------------------------------------------------------------------
extern "C" void kernel_launch(void* const* d_in, const int* in_sizes, int n_in,
                              void* d_out, int out_size, void* d_ws, size_t ws_size,
                              hipStream_t stream) {
  const float* x    = (const float*)d_in[0];   // (64, 2048, 256)
  const float* Wk   = (const float*)d_in[1];   // (256, 768)
  const float* Uk   = (const float*)d_in[2];   // (256, 768)
  const float* bias = (const float*)d_in[3];   // (768,)
  float* out = (float*)d_out;
  char*  ws  = (char*)d_ws;

  char* ufrag = ws + UFRAG_OFF;
  char* wfrag = ws + WFRAG_OFF;
  char* xws   = ws + XWS_OFF;

  (void)in_sizes; (void)n_in; (void)out_size; (void)ws_size;

  (void)hipFuncSetAttribute((const void*)xproj_kernel,
                            hipFuncAttributeMaxDynamicSharedMemorySize, 65536);
  (void)hipFuncSetAttribute((const void*)scan_kernel,
                            hipFuncAttributeMaxDynamicSharedMemorySize, 278528);

  // Phase 0: swizzle weights into bf16 WMMA B-fragment order
  swz_kernel<<<768, 256, 0, stream>>>(Wk, wfrag);
  swz_kernel<<<768, 256, 0, stream>>>(Uk, ufrag);

  // Phase 1: x_proj GEMM (bf16 WMMA), scan-fragment output layout
  xproj_kernel<<<1024, 256, 65536, stream>>>(x, bias, wfrag, xws);

  // Phase 2: persistent GRU scan, 4 WGs x 16 waves
  scan_kernel<<<4, 512, 278528, stream>>>(ufrag, xws, out);
}
